// RetroLLMTransformer_4114578669846
// MI455X (gfx1250) — matrix operand order repaired and checked by reading, not compiled, not run on hardware
//
#include <hip/hip_runtime.h>

// ---------------------------------------------------------------------------
// RetroLLM transformer forward for MI455X (gfx1250, wave32).
// - All GEMMs: v_wmma_f32_16x16x32_bf16, fp32 accumulation, single NT path
//   (weights pre-transposed once during fp32->bf16 conversion).
// - K-loop staging: Tensor Data Mover (tensor_load_to_lds) into double-
//   buffered LDS, handed off via s_wait_tensorcnt + workgroup barrier.
// - 2-phase unrolled pipeline: buffer indices are compile-time constants.
// - Wave tile 64x64: 16 WMMA per 16 ds_load_b128 per K-step.
// ---------------------------------------------------------------------------

typedef __bf16 bf16_t;
typedef __bf16 v16bf __attribute__((ext_vector_type(16)));
typedef float  v8f   __attribute__((ext_vector_type(8)));
typedef unsigned int u32x4 __attribute__((ext_vector_type(4)));
typedef int          i32x4 __attribute__((ext_vector_type(4)));
typedef int          i32x8 __attribute__((ext_vector_type(8)));

static_assert(sizeof(bf16_t) == 2, "bf16 must be 2 bytes");

__device__ __forceinline__ bf16_t f2bf(float f) {
    unsigned u = __builtin_bit_cast(unsigned, f);
    u += 0x7FFFu + ((u >> 16) & 1u);
    unsigned short h = (unsigned short)(u >> 16);
    return __builtin_bit_cast(bf16_t, h);
}

// ---------------------------------------------------------------------------
// TDM: 2-D tile load (global -> LDS). D# packed per CDNA5 ISA ch.8.
// Issued per-wave (EXEC ignored); caller gates to one wave per workgroup.
// ---------------------------------------------------------------------------
__device__ __forceinline__ void tdm_load_2d(unsigned lds_off, unsigned long ga,
                                            unsigned tile_d0, unsigned tile_d1,
                                            unsigned tensor_d0, unsigned tensor_d1,
                                            unsigned long stride0) {
    u32x4 g0;
    g0[0] = 0x1u;                                   // count=1, user descriptor
    g0[1] = lds_off;                                // lds_addr (bytes)
    g0[2] = (unsigned)(ga & 0xFFFFFFFFu);           // global_addr[31:0]
    g0[3] = (unsigned)((ga >> 32) & 0x01FFFFFFu)    // global_addr[56:32]
          | (2u << 30);                             // type = 2 ("image")
    i32x8 g1;
    g1[0] = (int)(1u << 16);                        // data_size=1 => 2 bytes
    g1[1] = (int)((tensor_d0 & 0xFFFFu) << 16);     // tensor_dim0[15:0]
    g1[2] = (int)((tensor_d0 >> 16) | ((tensor_d1 & 0xFFFFu) << 16));
    g1[3] = (int)((tensor_d1 >> 16) | (tile_d0 << 16)); // tile_dim0
    g1[4] = (int)(tile_d1 & 0xFFFFu);               // tile_dim1 (tile_dim2=0)
    g1[5] = (int)(stride0 & 0xFFFFFFFFu);           // tensor_dim0_stride[31:0]
    g1[6] = (int)((stride0 >> 32) & 0xFFFFu);       // stride[47:32]
    g1[7] = 0;
    i32x4 z4 = {0, 0, 0, 0};
#if __clang_major__ >= 23
    i32x8 z8 = {0, 0, 0, 0, 0, 0, 0, 0};
    __builtin_amdgcn_tensor_load_to_lds(g0, g1, z4, z4, z8, 0);
#else
    __builtin_amdgcn_tensor_load_to_lds(g0, g1, z4, z4, 0);
#endif
}

__device__ __forceinline__ unsigned lds_offset_of(const void* p) {
    // generic LDS pointer: hardware LDS address is addr[31:0]
    return (unsigned)(unsigned long)p;
}

// ---------------------------------------------------------------------------
// fp32 -> bf16 bulk conversion (tok_embed stays [V,D]: already NT for LM head)
// ---------------------------------------------------------------------------
__global__ void f32_to_bf16_kernel(const float* __restrict__ in,
                                   bf16_t* __restrict__ out, long n) {
    long i = (long)blockIdx.x * blockDim.x + threadIdx.x;
    long stride = (long)gridDim.x * blockDim.x;
    for (; i < n; i += stride) out[i] = f2bf(in[i]);
}

// ---------------------------------------------------------------------------
// fp32 [K,N] -> bf16 [N,K] tiled transpose (weights, done once)
// ---------------------------------------------------------------------------
__global__ __launch_bounds__(256)
void f32_to_bf16_transpose_kernel(const float* __restrict__ in,
                                  bf16_t* __restrict__ out, int K, int N) {
    __shared__ float t[32][33];
    int nb = blockIdx.x * 32;
    int kb = blockIdx.y * 32;
    int tx = threadIdx.x & 31;
    int ty = threadIdx.x >> 5;
#pragma unroll
    for (int r = ty; r < 32; r += 8)
        t[r][tx] = in[(long)(kb + r) * N + (nb + tx)];
    __syncthreads();
#pragma unroll
    for (int r = ty; r < 32; r += 8)
        out[(long)(nb + r) * K + (kb + tx)] = f2bf(t[tx][r]);
}

// ---------------------------------------------------------------------------
// Embedding: X[row,:] = tok_embed[x[row]] + pos_embed[row % N]
// ---------------------------------------------------------------------------
__global__ void embed_kernel(const int* __restrict__ x,
                             const float* __restrict__ tokE,
                             const float* __restrict__ posE,
                             float* __restrict__ Xf,
                             bf16_t* __restrict__ Xb,
                             int N, int D) {
    int row = blockIdx.x;
    int n   = row % N;
    int tok = x[row];
    const float* te = tokE + (long)tok * D;
    const float* pe = posE + (long)n   * D;
    float*  xf = Xf + (long)row * D;
    bf16_t* xb = Xb + (long)row * D;
    for (int i = threadIdx.x; i < D; i += blockDim.x) {
        float v = te[i] + pe[i];
        xf[i] = v;
        xb[i] = f2bf(v);
    }
}

// ---------------------------------------------------------------------------
// NT WMMA GEMM with TDM-staged, double-buffered LDS tiles.
//   C = act(A @ B^T + bias) (+resid)
//   A : bf16 [M,K] row-major (+bsA per batch)
//   B : bf16 [N,K] row-major (+bsB per batch)
//   Cf: fp32 [M,N] | null;  Cb: bf16 [M,N] | null (CT: bf16 [N,M] transposed)
// Block 256 thr = 8 waves (2M x 4N), tile 128x256x32; wave tile 64x64.
// K must be a multiple of 64 (two pipeline phases per loop iteration).
// ---------------------------------------------------------------------------
#define BM 128
#define BNT 256
#define BK 32

template<bool RELU, bool CT>
__global__ __launch_bounds__(256)
void gemm_nt_wmma(const bf16_t* __restrict__ A,
                  const bf16_t* __restrict__ Bm,
                  const float*  __restrict__ bias,
                  const float*  __restrict__ resid,
                  float*  __restrict__ Cf,
                  bf16_t* __restrict__ Cb,
                  int M, int N, int K,
                  long bsA, long bsB, long bsC) {
    __shared__ bf16_t As[2][BM][BK];
    __shared__ bf16_t Bs[2][BNT][BK];

    const int tid   = threadIdx.x;
    const int wave  = tid >> 5;
    const int lane  = tid & 31;
    const int lhalf = lane >> 4;
    const int lmod  = lane & 15;

    const int n0 = blockIdx.x * BNT;
    const int m0 = blockIdx.y * BM;
    const int bz = blockIdx.z;

    const bf16_t* Ab = A  + (long)bz * bsA;
    const bf16_t* Bb = Bm + (long)bz * bsB;

    // 8 waves: 2 in M x 4 in N; wave tile 64 (M) x 64 (N)
    const int wm = (wave >> 2) * 64;
    const int wn = (wave & 3)  * 64;

    float bv[4];
#pragma unroll
    for (int nt = 0; nt < 4; ++nt) {
        int n = n0 + wn + nt * 16 + lmod;
        bv[nt] = bias ? bias[n] : 0.0f;
    }
    v8f acc[4][4];
#pragma unroll
    for (int mt = 0; mt < 4; ++mt)
#pragma unroll
        for (int nt = 0; nt < 4; ++nt)
#pragma unroll
            for (int r = 0; r < 8; ++r) acc[mt][nt][r] = bv[nt];

    // byte addresses of this block's K=0 tiles + constant LDS buffer offsets
    const unsigned long gA = (unsigned long)(Ab + (long)m0 * K);
    const unsigned long gB = (unsigned long)(Bb + (long)n0 * K);
    const unsigned ldsA[2] = { lds_offset_of(&As[0][0][0]),
                               lds_offset_of(&As[1][0][0]) };
    const unsigned ldsB[2] = { lds_offset_of(&Bs[0][0][0]),
                               lds_offset_of(&Bs[1][0][0]) };

    // one K-step of compute from a given (compile-time) buffer
    auto compute_tile = [&](const bf16_t (*Asb)[BK], const bf16_t (*Bsb)[BK]) {
        const int kbA = lhalf * 8;        // A: lanes 0-15 K-base 0, 16-31 base 8
        v16bf afrag[4];
#pragma unroll
        for (int mt = 0; mt < 4; ++mt) {
            int m = wm + mt * 16 + lmod;
            union { v16bf v; uint4 q[2]; } fa;
            fa.q[0] = *reinterpret_cast<const uint4*>(&Asb[m][kbA]);
            fa.q[1] = *reinterpret_cast<const uint4*>(&Asb[m][16 + kbA]);
            afrag[mt] = fa.v;
        }
        const int kbB = lhalf * 8;        // B: lanes 0-15 K-base 0, 16-31 base 8
        v16bf bfrag[4];
#pragma unroll
        for (int nt = 0; nt < 4; ++nt) {
            int n = wn + nt * 16 + lmod;
            union { v16bf v; uint4 q[2]; } fb;
            fb.q[0] = *reinterpret_cast<const uint4*>(&Bsb[n][kbB]);
            fb.q[1] = *reinterpret_cast<const uint4*>(&Bsb[n][16 + kbB]);
            bfrag[nt] = fb.v;
        }
#pragma unroll
        for (int mt = 0; mt < 4; ++mt)
#pragma unroll
            for (int nt = 0; nt < 4; ++nt)
                acc[mt][nt] = __builtin_amdgcn_wmma_f32_16x16x32_bf16(
                    false, afrag[mt], false, bfrag[nt],
                    (short)0, acc[mt][nt], false, false);
    };

    // prologue: wave 0 issues TDM for buffer 0 (tile at k=0)
    if (wave == 0) {
        tdm_load_2d(ldsA[0], gA, BK, BM,  (unsigned)K, (unsigned)M, (unsigned long)K);
        tdm_load_2d(ldsB[0], gB, BK, BNT, (unsigned)K, (unsigned)N, (unsigned long)K);
    }

    // two phases per iteration; CUR/NXT are compile-time constants
#define PIPE_STEP(CUR, NXT, K0)                                                 \
    do {                                                                        \
        if (wave == 0) __builtin_amdgcn_s_wait_tensorcnt(0);                    \
        __syncthreads();                                                        \
        if (wave == 0 && (K0) + BK < K) {                                       \
            tdm_load_2d(ldsA[NXT], gA + 2ul * (unsigned long)((K0) + BK),       \
                        BK, BM,  (unsigned)K, (unsigned)M, (unsigned long)K);   \
            tdm_load_2d(ldsB[NXT], gB + 2ul * (unsigned long)((K0) + BK),       \
                        BK, BNT, (unsigned)K, (unsigned)N, (unsigned long)K);   \
        }                                                                       \
        compute_tile(As[CUR], Bs[CUR]);                                         \
        __syncthreads();                                                        \
    } while (0)

    for (int k0 = 0; k0 < K; k0 += 2 * BK) {
        PIPE_STEP(0, 1, k0);
        PIPE_STEP(1, 0, k0 + BK);
    }
#undef PIPE_STEP

    // epilogue: C/D layout VGPR r -> M = 16*mt + r + 8*lhalf, N = lane&15
    const float* rb = resid ? resid + (long)bz * bsC : nullptr;
    float*  cf = Cf ? Cf + (long)bz * bsC : nullptr;
    bf16_t* cb = Cb ? Cb + (long)bz * bsC : nullptr;
#pragma unroll
    for (int mt = 0; mt < 4; ++mt) {
#pragma unroll
        for (int nt = 0; nt < 4; ++nt) {
            int n = n0 + wn + nt * 16 + lmod;
#pragma unroll
            for (int r = 0; r < 8; ++r) {
                int m = m0 + wm + mt * 16 + r + lhalf * 8;
                float v = acc[mt][nt][r];
                if (RELU) v = v > 0.0f ? v : 0.0f;
                if (rb) v += rb[(long)m * N + n];
                if (cf) cf[(long)m * N + n] = v;
                if (cb) {
                    if (CT) cb[(long)n * M + m] = f2bf(v);   // transposed store
                    else    cb[(long)m * N + n] = f2bf(v);
                }
            }
        }
    }
}

// ---------------------------------------------------------------------------
// Causal softmax, 1/sqrt(D) scale folded in, bf16 weights out
// ---------------------------------------------------------------------------
__global__ __launch_bounds__(256)
void softmax_causal_kernel(const float* __restrict__ scores,
                           bf16_t* __restrict__ Wout, int N, float scale) {
    long row = blockIdx.x;               // b*N + q
    int  q   = (int)(row % N);
    const float* s = scores + row * (long)N;
    bf16_t* w = Wout + row * (long)N;
    __shared__ float red[256];
    int tid = threadIdx.x;

    float m = -3.402823466e38f;
    for (int k = tid; k <= q; k += 256) m = fmaxf(m, s[k] * scale);
    red[tid] = m; __syncthreads();
    for (int off = 128; off > 0; off >>= 1) {
        if (tid < off) red[tid] = fmaxf(red[tid], red[tid + off]);
        __syncthreads();
    }
    m = red[0]; __syncthreads();

    float sum = 0.0f;
    for (int k = tid; k <= q; k += 256) sum += __expf(s[k] * scale - m);
    red[tid] = sum; __syncthreads();
    for (int off = 128; off > 0; off >>= 1) {
        if (tid < off) red[tid] += red[tid + off];
        __syncthreads();
    }
    float inv = 1.0f / red[0];

    for (int k = tid; k < N; k += 256) {
        float v = (k <= q) ? __expf(s[k] * scale - m) * inv : 0.0f;
        w[k] = f2bf(v);
    }
}

// ---------------------------------------------------------------------------
// Host-side orchestration
// ---------------------------------------------------------------------------
extern "C" void kernel_launch(void* const* d_in, const int* in_sizes, int n_in,
                              void* d_out, int out_size, void* d_ws, size_t ws_size,
                              hipStream_t stream) {
    (void)in_sizes; (void)n_in; (void)out_size; (void)ws_size;

    const int Bb = 2, N = 2048, D = 1024, FF = 4096, V = 32000;
    const int M = Bb * N;                 // 4096 token rows

    const int*   x    = (const int*)  d_in[0];
    const float* tokE = (const float*)d_in[1];
    const float* posE = (const float*)d_in[2];
    const float* Wq   = (const float*)d_in[3];
    const float* bq   = (const float*)d_in[4];
    const float* Wk   = (const float*)d_in[5];
    const float* bk   = (const float*)d_in[6];
    const float* Wv   = (const float*)d_in[7];
    const float* bv   = (const float*)d_in[8];
    const float* Wo   = (const float*)d_in[9];
    const float* bo   = (const float*)d_in[10];
    const float* W1   = (const float*)d_in[11];
    const float* b1   = (const float*)d_in[12];
    const float* W2   = (const float*)d_in[13];
    const float* b2   = (const float*)d_in[14];
    const float* lmb  = (const float*)d_in[15];
    float* logits = (float*)d_out;

    size_t off = 0;
    auto alloc = [&](size_t bytes) -> char* {
        char* p = (char*)d_ws + off;
        off += (bytes + 255) & ~(size_t)255;
        return p;
    };
    float*  Xf    = (float*) alloc((size_t)M * D * 4);
    bf16_t* Xb    = (bf16_t*)alloc((size_t)M * D * 2);
    bf16_t* Wqt   = (bf16_t*)alloc((size_t)D * D * 2);
    bf16_t* Wkt   = (bf16_t*)alloc((size_t)D * D * 2);
    bf16_t* Wvt   = (bf16_t*)alloc((size_t)D * D * 2);
    bf16_t* Wot   = (bf16_t*)alloc((size_t)D * D * 2);
    bf16_t* W1t   = (bf16_t*)alloc((size_t)D * FF * 2);
    bf16_t* W2t   = (bf16_t*)alloc((size_t)FF * D * 2);
    bf16_t* tokEb = (bf16_t*)alloc((size_t)V * D * 2);
    bf16_t* Qb    = (bf16_t*)alloc((size_t)M * D * 2);
    bf16_t* Kb    = (bf16_t*)alloc((size_t)M * D * 2);
    bf16_t* Vt    = (bf16_t*)alloc((size_t)M * D * 2);   // per-batch [D,N]
    float*  Sc    = (float*) alloc((size_t)Bb * N * N * 4);
    bf16_t* Wsm   = (bf16_t*)alloc((size_t)Bb * N * N * 2);
    bf16_t* AOb   = (bf16_t*)alloc((size_t)M * D * 2);
    float*  X2f   = (float*) alloc((size_t)M * D * 4);
    bf16_t* X2b   = (bf16_t*)alloc((size_t)M * D * 2);
    bf16_t* Hb    = (bf16_t*)alloc((size_t)M * FF * 2);
    bf16_t* X3b   = (bf16_t*)alloc((size_t)M * D * 2);
    float*  X3f   = (float*) alloc((size_t)M * D * 4);

    // 1) weight conversions: fp32 [K,N] -> bf16 [N,K] (transpose once)
    auto cvtT = [&](const float* src, bf16_t* dst, int K_, int N_) {
        dim3 g(N_ / 32, K_ / 32, 1);
        f32_to_bf16_transpose_kernel<<<g, 256, 0, stream>>>(src, dst, K_, N_);
    };
    cvtT(Wq, Wqt, D, D);
    cvtT(Wk, Wkt, D, D);
    cvtT(Wv, Wvt, D, D);
    cvtT(Wo, Wot, D, D);
    cvtT(W1, W1t, D, FF);
    cvtT(W2, W2t, FF, D);
    f32_to_bf16_kernel<<<2048, 256, 0, stream>>>(tokE, tokEb, (long)V * D);

    // 2) embedding
    embed_kernel<<<M, 256, 0, stream>>>(x, tokE, posE, Xf, Xb, N, D);

    const long sND = (long)N * D;
    const long sNN = (long)N * N;

    // 3) Q,K projections
    {
        dim3 g(D / BNT, M / BM, 1);
        gemm_nt_wmma<false, false><<<g, 256, 0, stream>>>(
            Xb, Wqt, bq, nullptr, nullptr, Qb, M, D, D, 0, 0, 0);
        gemm_nt_wmma<false, false><<<g, 256, 0, stream>>>(
            Xb, Wkt, bk, nullptr, nullptr, Kb, M, D, D, 0, 0, 0);
    }
    // V projection, stored transposed per batch: Vt[b] = (X Wv)^T [D,N]
    {
        dim3 g(D / BNT, N / BM, Bb);
        gemm_nt_wmma<false, true><<<g, 256, 0, stream>>>(
            Xb, Wvt, bv, nullptr, nullptr, Vt, N, D, D, sND, 0, sND);
    }

    // 4) scores = Q @ K^T per batch
    {
        dim3 g(N / BNT, N / BM, Bb);
        gemm_nt_wmma<false, false><<<g, 256, 0, stream>>>(
            Qb, Kb, nullptr, nullptr, Sc, nullptr, N, N, D, sND, sND, sNN);
    }

    // 5) causal softmax (scale 1/sqrt(1024))
    softmax_causal_kernel<<<Bb * N, 256, 0, stream>>>(Sc, Wsm, N, 1.0f / 32.0f);

    // 6) attn = weights @ (Vt)^T per batch
    {
        dim3 g(D / BNT, N / BM, Bb);
        gemm_nt_wmma<false, false><<<g, 256, 0, stream>>>(
            Wsm, Vt, nullptr, nullptr, nullptr, AOb, N, D, N, sNN, sND, sND);
    }

    // 7) X2 = X + attn @ Wo + bo
    {
        dim3 g(D / BNT, M / BM, 1);
        gemm_nt_wmma<false, false><<<g, 256, 0, stream>>>(
            AOb, Wot, bo, Xf, X2f, X2b, M, D, D, 0, 0, 0);
    }

    // 8) H = relu(X2 @ W1 + b1)
    {
        dim3 g(FF / BNT, M / BM, 1);
        gemm_nt_wmma<true, false><<<g, 256, 0, stream>>>(
            X2b, W1t, b1, nullptr, nullptr, Hb, M, FF, D, 0, 0, 0);
    }

    // 9) X3 = X2 + H @ W2 + b2
    {
        dim3 g(D / BNT, M / BM, 1);
        gemm_nt_wmma<false, false><<<g, 256, 0, stream>>>(
            Hb, W2t, b2, X2f, X3f, X3b, M, D, FF, 0, 0, 0);
    }

    // 10) logits = X3 @ tok_embed^T + lm_bias -> fp32 d_out
    {
        dim3 g(V / BNT, M / BM, 1);
        gemm_nt_wmma<false, false><<<g, 256, 0, stream>>>(
            X3b, tokEb, lmb, nullptr, logits, nullptr, M, V, D, 0, 0, 0);
    }
}
